// CustomGAT_27693949125005
// MI455X (gfx1250) — matrix operand order repaired and checked by reading, not compiled, not run on hardware
//
#include <hip/hip_runtime.h>
#include <hip/hip_bf16.h>

// ---------------- problem constants (from reference) ----------------
#define N_PANO 50000
#define N_FP   10000
#define D_IN   128
#define D_FP   64
#define HID    128
#define OUT_C  64
#define E_PP   800000
#define E_PF   50000
#define MP     50016   // N_PANO padded to multiple of 32 rows
#define KB     4       // K=128 -> 4 steps of 32

typedef __attribute__((ext_vector_type(16))) __bf16 v16bf;
typedef __attribute__((ext_vector_type(8)))  float  v8f;

// f32 -> bf16 round-to-nearest-even (bit-level, stored as ushort)
__device__ inline unsigned short f2bf(float f) {
    unsigned int u = __float_as_uint(f);
    u += 0x7FFFu + ((u >> 16) & 1u);
    return (unsigned short)(u >> 16);
}

// ---------------------------------------------------------------------------
// Pre-swizzle a weight matrix W[K=128][N] (f32, row-major) into fragment-ready
// bf16: layout ((tile*KB + kb)*32 + lane)*16 + e, where for the B operand of
// v_wmma_f32_16x16x32_bf16:  col = tile*16 + (lane&15),  k = kb*32 + (lane>>4)*16 + e
// ---------------------------------------------------------------------------
__global__ void prep_wfrag(const float* __restrict__ W, int N,
                           unsigned short* __restrict__ Wf) {
    int idx = blockIdx.x * blockDim.x + threadIdx.x;
    int total = (N >> 4) * KB * 32 * 16;
    if (idx >= total) return;
    int e    = idx & 15;
    int lane = (idx >> 4) & 31;
    int kb   = (idx >> 9) & 3;
    int t    = idx >> 11;
    int k    = kb * 32 + (lane >> 4) * 16 + e;
    int col  = t * 16 + (lane & 15);
    Wf[idx] = f2bf(W[k * N + col]);
}

// ---------------------------------------------------------------------------
// Fused GEMM: O0 = A @ W0 [, O1 = A @ W1 + bias1].  A is f32 [Mpad x 128].
// Wf holds the fragment-swizzled bf16 weights for (TPW*4) 16-col tiles.
// Block = 256 threads = 8 waves; 32 rows per block; each wave: 16-row strip,
// TPW column tiles, K-loop of 4 x v_wmma_f32_16x16x32_bf16.
// ---------------------------------------------------------------------------
template <int TPW>
__global__ void __launch_bounds__(256)
wmma_gemm_k128(const float* __restrict__ A,
               const unsigned short* __restrict__ Wf,
               int N0tiles, const float* __restrict__ bias1,
               float* __restrict__ O0, int ld0,
               float* __restrict__ O1, int ld1) {
    __shared__ __align__(32) unsigned short As[2 * KB * 32 * 16]; // 8 KB

    const int tid  = threadIdx.x;
    const int row0 = blockIdx.x * 32;

    // ---- stage A tile (32 rows x 128 cols) as fragment-ready bf16 in LDS ----
    {
        int lane  = tid & 31;
        int kb    = (tid >> 5) & 3;
        int strip = tid >> 7;
        int half  = lane >> 4;
        int m     = lane & 15;
        const float* arow = A + (size_t)(row0 + strip * 16 + m) * 128 + kb * 32;
        unsigned short* dst = &As[((strip * KB + kb) * 32 + lane) * 16];
#pragma unroll
        for (int v = 0; v < 8; ++v) {
            int kloc = half * 16 + v * 2;
            float2 f = *(const float2*)(arow + kloc);
            dst[2 * v]     = f2bf(f.x);
            dst[2 * v + 1] = f2bf(f.y);
        }
    }
    __syncthreads();

    const int wave  = tid >> 5;
    const int lane  = tid & 31;
    const int strip = wave & 1;     // 16-row strip within the 32-row block tile
    const int tbase = wave >> 1;    // 0..3

    v8f acc[TPW] = {};
#pragma unroll
    for (int kb = 0; kb < KB; ++kb) {
        v16bf a = *(const v16bf*)&As[((strip * KB + kb) * 32 + lane) * 16];
#pragma unroll
        for (int i = 0; i < TPW; ++i) {
            int t = tbase + i * 4;
            v16bf b = *(const v16bf*)&Wf[((size_t)(t * KB + kb) * 32 + lane) * 16];
            acc[i] = __builtin_amdgcn_wmma_f32_16x16x32_bf16(
                false, a, false, b, (short)0, acc[i], false, false);
        }
    }

    // ---- store C (layout: VGPR r -> row r + 8*half, col = lane&15) ----
    const int half = lane >> 4;
    const int n    = lane & 15;
    const int rowb = row0 + strip * 16 + half * 8;
#pragma unroll
    for (int i = 0; i < TPW; ++i) {
        int t = tbase + i * 4;
        float* O; int col0, ld; float badd = 0.0f;
        if (t < N0tiles) { O = O0; col0 = t * 16; ld = ld0; }
        else             { O = O1; col0 = (t - N0tiles) * 16; ld = ld1;
                           badd = bias1[col0 + n]; }
#pragma unroll
        for (int r = 0; r < 8; ++r)
            O[(size_t)(rowb + r) * ld + col0 + n] = acc[i][r] + badd;
    }
}

// wvec[k] = sum_n W[k][n] * a[n]   (folds the Wd GEMM into one vector)
__global__ void weight_vec(const float* __restrict__ W, const float* __restrict__ a,
                           float* __restrict__ out, int K, int N) {
    int k = blockIdx.x * blockDim.x + threadIdx.x;
    if (k >= K) return;
    float s = 0.f;
    for (int nn = 0; nn < N; ++nn) s += W[k * N + nn] * a[nn];
    out[k] = s;
}

// out[i] = dot(X[i, :D], v)  — one wave32 per row, lane-strided + shfl reduce
__global__ void __launch_bounds__(256)
row_dot(const float* __restrict__ X, int ld, int D,
        const float* __restrict__ v, float* __restrict__ out, int M) {
    int w    = (blockIdx.x * blockDim.x + threadIdx.x) >> 5;
    int lane = threadIdx.x & 31;
    if (w >= M) return;
    const float* xr = X + (size_t)w * ld;
    float s = 0.f;
    for (int c = lane; c < D; c += 32) s += xr[c] * v[c];
#pragma unroll
    for (int off = 16; off; off >>= 1) s += __shfl_xor(s, off, 32);
    if (lane == 0) out[w] = s;
}

__global__ void copyf(const float* __restrict__ a, float* __restrict__ b, int n) {
    int i = blockIdx.x * blockDim.x + threadIdx.x;
    if (i < n) b[i] = a[i];
}

__global__ void init_softmax(float* __restrict__ m, float* __restrict__ den, int n) {
    int i = blockIdx.x * blockDim.x + threadIdx.x;
    if (i < n) { m[i] = -3.0e38f; den[i] = 0.f; }
}

__global__ void init_rows_bias(float* __restrict__ O, const float* __restrict__ b,
                               int n, int C) {
    int i = blockIdx.x * blockDim.x + threadIdx.x;
    if (i < n * C) O[i] = b[i % C];
}

__device__ inline void atomicMaxF(float* addr, float val) {
    float cur = __int_as_float(__hip_atomic_load((int*)addr, __ATOMIC_RELAXED,
                                                 __HIP_MEMORY_SCOPE_AGENT));
    while (cur < val) {
        int curi = __float_as_int(cur);
        int prev = atomicCAS((int*)addr, curi, __float_as_int(val));
        if (prev == curi) break;
        cur = __int_as_float(prev);
    }
}

__global__ void edge_logits(const int* __restrict__ src, const int* __restrict__ dst,
                            const float* __restrict__ es, const float* __restrict__ ed,
                            float* __restrict__ ev, float* __restrict__ m, int E) {
    int i = blockIdx.x * blockDim.x + threadIdx.x;
    if (i >= E) return;
    float v = es[src[i]] + ed[dst[i]];
    v = v > 0.f ? v : 0.2f * v;          // leaky_relu(0.2)
    ev[i] = v;
    atomicMaxF(&m[dst[i]], v);
}

__global__ void edge_exp(const int* __restrict__ dst, const float* __restrict__ m,
                         float* __restrict__ ev, float* __restrict__ den, int E) {
    int i = blockIdx.x * blockDim.x + threadIdx.x;
    if (i >= E) return;
    int d = dst[i];
    float x = __expf(ev[i] - m[d]);
    ev[i] = x;
    atomicAdd(&den[d], x);
}

// one wave per edge: agg[dst] += alpha * H[src]  (lane-strided, coalesced)
__global__ void __launch_bounds__(256)
edge_scatter(const int* __restrict__ src, const int* __restrict__ dst,
             const float* __restrict__ ev, const float* __restrict__ den,
             const float* __restrict__ H, float* __restrict__ agg, int E, int C) {
    int gw   = (blockIdx.x * blockDim.x + threadIdx.x) >> 5;
    int lane = threadIdx.x & 31;
    int nw   = (gridDim.x * blockDim.x) >> 5;
    for (int e = gw; e < E; e += nw) {
        int s = src[e], d = dst[e];
        float alpha = ev[e] / den[d];
        const float* hrow = H + (size_t)s * C;
        float* orow = agg + (size_t)d * C;
        for (int c = lane; c < C; c += 32)
            atomicAdd(&orow[c], alpha * hrow[c]);
    }
}

__global__ void post_relu(const float* __restrict__ agg, const float* __restrict__ lin,
                          float* __restrict__ h, int total) {
    int i = blockIdx.x * blockDim.x + threadIdx.x;
    if (i < total) { float x = agg[i] + lin[i]; h[i] = x > 0.f ? x : 0.f; }
}

// ---------------------------------------------------------------------------
extern "C" void kernel_launch(void* const* d_in, const int* in_sizes, int n_in,
                              void* d_out, int out_size, void* d_ws, size_t ws_size,
                              hipStream_t stream) {
    (void)in_sizes; (void)n_in; (void)out_size; (void)ws_size;
    const float* x_pano = (const float*)d_in[0];
    const float* x_fp   = (const float*)d_in[1];
    const float* Ws0 = (const float*)d_in[2];
    const float* Wd0 = (const float*)d_in[3];
    const float* as0 = (const float*)d_in[4];
    const float* ad0 = (const float*)d_in[5];
    const float* b0  = (const float*)d_in[6];
    const float* Lw0 = (const float*)d_in[7];
    const float* Lb0 = (const float*)d_in[8];
    const float* Ws1 = (const float*)d_in[9];
    const float* Wd1 = (const float*)d_in[10];
    const float* as1 = (const float*)d_in[11];
    const float* ad1 = (const float*)d_in[12];
    const float* b1  = (const float*)d_in[13];
    const float* Lw1 = (const float*)d_in[14];
    const float* Lb1 = (const float*)d_in[15];
    const float* Wts = (const float*)d_in[16];
    const float* Wtd = (const float*)d_in[17];
    const float* ats = (const float*)d_in[18];
    const float* atd = (const float*)d_in[19];
    const float* bt  = (const float*)d_in[20];
    const int* pp_src = (const int*)d_in[21];
    const int* pp_dst = pp_src + E_PP;
    const int* pf_src = (const int*)d_in[22];
    const int* pf_dst = (const int*)d_in[23];

    // ---- carve workspace ----
    char* base = (char*)d_ws;
    size_t off = 0;
    auto carve = [&](size_t bytes) -> void* {
        off = (off + 255) & ~(size_t)255;
        void* p = base + off;
        off += bytes;
        return p;
    };
    float* h    = (float*)carve((size_t)MP * 128 * 4);
    float* hs   = (float*)carve((size_t)MP * 128 * 4);
    float* lin  = (float*)carve((size_t)MP * 128 * 4);
    float* agg  = (float*)carve((size_t)MP * 128 * 4);
    float* es   = (float*)carve((size_t)MP * 4);
    float* ed   = (float*)carve((size_t)MP * 4);
    float* mbuf = (float*)carve((size_t)MP * 4);
    float* den  = (float*)carve((size_t)MP * 4);
    float* ev   = (float*)carve((size_t)E_PP * 4);
    float* wvec = (float*)carve(128 * 4);
    unsigned short* Wf0 = (unsigned short*)carve(16 * 2048 * 2);
    unsigned short* Wf1 = (unsigned short*)carve(16 * 2048 * 2);
    unsigned short* Wft = (unsigned short*)carve(4 * 2048 * 2);

    // ---- pre-swizzle weights to fragment-ready bf16 (every call; tiny) ----
    prep_wfrag<<<64, 256, 0, stream>>>(Ws0, 128, Wf0);
    prep_wfrag<<<64, 256, 0, stream>>>(Lw0, 128, Wf0 + 8 * 2048);
    prep_wfrag<<<64, 256, 0, stream>>>(Ws1, 128, Wf1);
    prep_wfrag<<<64, 256, 0, stream>>>(Lw1, 128, Wf1 + 8 * 2048);
    prep_wfrag<<<32, 256, 0, stream>>>(Wts, 64, Wft);

    copyf<<<(N_PANO * 128 + 255) / 256, 256, 0, stream>>>(x_pano, h, N_PANO * 128);

    const float* WdL[2] = {Wd0, Wd1};
    const float* adL[2] = {ad0, ad1};
    const float* asL[2] = {as0, as1};
    const float* bL[2]  = {b0, b1};
    const float* LbL[2] = {Lb0, Lb1};
    unsigned short* WfL[2] = {Wf0, Wf1};

    for (int l = 0; l < 2; ++l) {
        // hs = h@Ws ; lin = h@Lw + Lb   (fused, 16 WMMA col-tiles)
        wmma_gemm_k128<4><<<MP / 32, 256, 0, stream>>>(
            h, WfL[l], 8, LbL[l], hs, 128, lin, 128);
        // es = hs . a_s ; ed = h . (Wd a_d)
        row_dot<<<(N_PANO * 32 + 255) / 256, 256, 0, stream>>>(hs, 128, 128, asL[l], es, N_PANO);
        weight_vec<<<1, 128, 0, stream>>>(WdL[l], adL[l], wvec, 128, 128);
        row_dot<<<(N_PANO * 32 + 255) / 256, 256, 0, stream>>>(h, 128, 128, wvec, ed, N_PANO);
        // segment softmax + aggregation
        init_softmax<<<(N_PANO + 255) / 256, 256, 0, stream>>>(mbuf, den, N_PANO);
        init_rows_bias<<<(MP * 128 + 255) / 256, 256, 0, stream>>>(agg, bL[l], MP, 128);
        edge_logits<<<(E_PP + 255) / 256, 256, 0, stream>>>(pp_src, pp_dst, es, ed, ev, mbuf, E_PP);
        edge_exp<<<(E_PP + 255) / 256, 256, 0, stream>>>(pp_dst, mbuf, ev, den, E_PP);
        edge_scatter<<<12500, 256, 0, stream>>>(pp_src, pp_dst, ev, den, hs, agg, E_PP, 128);
        // h = relu(conv + lin)
        post_relu<<<(MP * 128 + 255) / 256, 256, 0, stream>>>(agg, lin, h, MP * 128);
    }

    // ---- translate layer: pano -> footprint (OUT=64), writes d_out ----
    float* out_fp = (float*)d_out;
    wmma_gemm_k128<1><<<MP / 32, 256, 0, stream>>>(h, Wft, 4, nullptr, hs, 64, nullptr, 0);
    row_dot<<<(N_PANO * 32 + 255) / 256, 256, 0, stream>>>(hs, 64, 64, ats, es, N_PANO);
    weight_vec<<<1, 64, 0, stream>>>(Wtd, atd, wvec, 64, 64);
    row_dot<<<(N_FP * 32 + 255) / 256, 256, 0, stream>>>(x_fp, 64, 64, wvec, ed, N_FP);
    init_softmax<<<(N_FP + 255) / 256, 256, 0, stream>>>(mbuf, den, N_FP);
    init_rows_bias<<<(N_FP * 64 + 255) / 256, 256, 0, stream>>>(out_fp, bt, N_FP, 64);
    edge_logits<<<(E_PF + 255) / 256, 256, 0, stream>>>(pf_src, pf_dst, es, ed, ev, mbuf, E_PF);
    edge_exp<<<(E_PF + 255) / 256, 256, 0, stream>>>(pf_dst, mbuf, ev, den, E_PF);
    edge_scatter<<<1563, 256, 0, stream>>>(pf_src, pf_dst, ev, den, hs, out_fp, E_PF, 64);
}
